// FeedForward_996432413508
// MI455X (gfx1250) — matrix operand order repaired and checked by reading, not compiled, not run on hardware
//
#include <hip/hip_runtime.h>
#include <hip/hip_bf16.h>

typedef __attribute__((ext_vector_type(16))) __bf16 v16bf;
typedef __attribute__((ext_vector_type(8)))  __bf16 v8bf;
typedef __attribute__((ext_vector_type(8)))  float  v8f;
typedef __attribute__((ext_vector_type(4))) unsigned int u32x4;
typedef __attribute__((ext_vector_type(8))) int i32x8;
typedef __attribute__((ext_vector_type(4))) int i32x4;

#define BETA_C 0.3f
#define THR_C  0.2f

// Block tile geometry: 8 waves, each 64x64 -> block = 128(M) x 256(N).
// K step = 32. LDS rows padded by TDM: 32 bf16 (16 DW) + 4 DW pad = 40 bf16 / 80 B.
#define LDS_ROW   40            // bf16 elements per padded LDS row
#define A_ROWS    128
#define B_ROWS    256
#define A_STAGE   (A_ROWS * LDS_ROW)              // 5120  elems (10240 B)
#define B_STAGE   (B_ROWS * LDS_ROW)              // 10240 elems (20480 B)
#define STAGE     (A_STAGE + B_STAGE)             // 15360 elems (30720 B)

// ---------------------------------------------------------------------------
// Prep: token shift + mix, fp32 -> bf16
// ---------------------------------------------------------------------------
__global__ void prep_xs_kernel(const float* __restrict__ x,
                               const float* __restrict__ mix,
                               __bf16* __restrict__ xs,
                               int S, int D, long total) {
    long i = (long)blockIdx.x * blockDim.x + threadIdx.x;
    if (i >= total) return;
    int  d = (int)(i % D);
    long t = i / D;
    int  s = (int)(t % S);
    float m  = mix[d];
    float xv = x[i];
    float pv = (s == 0) ? 0.f : x[i - D];
    xs[i] = (__bf16)(xv * m + pv * (1.f - m));
}

__global__ void f32_to_bf16_kernel(const float* __restrict__ in,
                                   __bf16* __restrict__ out, long n) {
    long i = (long)blockIdx.x * blockDim.x + threadIdx.x;
    if (i < n) out[i] = (__bf16)in[i];
}

// ---------------------------------------------------------------------------
// Tensor Data Mover: DMA a [rows x 32] bf16 tile (row stride = K elements)
// from global into LDS at lds_byte, padding each 64B row with 16B (-> 80B rows).
// Descriptor layout per CDNA5 ISA 8.3/8.4 (D# group0 + group1; groups 2/3 and
// the trailing group zero for a 2D tile). 6-arg builtin form (clang-23).
// ---------------------------------------------------------------------------
__device__ __forceinline__ void tdm_load_tile(const __bf16* gptr,
                                              unsigned lds_byte,
                                              unsigned rows,
                                              unsigned K) {
    unsigned long long ga = (unsigned long long)gptr;
    u32x4 g0;
    g0[0] = 1u;                                            // count=1, user mode
    g0[1] = lds_byte;                                      // lds_addr
    g0[2] = (unsigned)(ga & 0xffffffffu);                  // global_addr[31:0]
    g0[3] = (unsigned)((ga >> 32) & 0x01ffffffu)           // global_addr[56:32]
          | (2u << 30);                                    // type=2 (image)

    // group1 bitfield (256b) built as 4x u64:
    //  [17:16] data_size=1 (2B)  [20] pad_enable  [24:22] pad_interval=3 (16 DW)
    //  [31:25] pad_amount=3 (4 DW)  [79:48] tensor_dim0=K  [111:80] tensor_dim1=rows
    //  [127:112] tile_dim0=32  [143:128] tile_dim1=rows  [207:160] dim0_stride=K
    unsigned long long flags = (1ull << 16) | (1ull << 20) | (3ull << 22) | (3ull << 25);
    unsigned long long q0 = flags | ((unsigned long long)(K & 0xffffu) << 48);
    unsigned long long q1 = (unsigned long long)(K >> 16)
                          | ((unsigned long long)rows << 16)
                          | (32ull << 48);
    unsigned long long q2 = (unsigned long long)(rows & 0xffffu)
                          | ((unsigned long long)K << 32);
    unsigned long long q3 = 0ull;                          // stride hi / dim1_stride
    i32x8 g1;
    g1[0] = (int)(unsigned)q0; g1[1] = (int)(unsigned)(q0 >> 32);
    g1[2] = (int)(unsigned)q1; g1[3] = (int)(unsigned)(q1 >> 32);
    g1[4] = (int)(unsigned)q2; g1[5] = (int)(unsigned)(q2 >> 32);
    g1[6] = (int)(unsigned)q3; g1[7] = (int)(unsigned)(q3 >> 32);
    i32x4 g2 = {0, 0, 0, 0};
    i32x4 g3 = {0, 0, 0, 0};
    i32x8 g4 = {0, 0, 0, 0, 0, 0, 0, 0};
    __builtin_amdgcn_tensor_load_to_lds(g0, g1, g2, g3, g4, 0);
}

// ---------------------------------------------------------------------------
// WMMA GEMM with TDM->LDS double-buffered staging.
//   out[m,n] = epilogue( sum_k A[m,k] * W[n,k] ), A:[M,K], W:[N,K] bf16 row-major
//   MODE 0: relu(acc)^2 -> bf16   MODE 1: sigmoid -> bf16   MODE 2: *gate -> f32
// ---------------------------------------------------------------------------
template <int MODE>
__global__ __launch_bounds__(256)
void wmma_gemm_kernel(const __bf16* __restrict__ A,
                      const __bf16* __restrict__ W,
                      int M, int N, int K,
                      __bf16* __restrict__ outB,
                      float*  __restrict__ outF,
                      const __bf16* __restrict__ gate) {
    __shared__ __bf16 smem[2 * STAGE];                     // 61440 B

    const int lane  = threadIdx.x & 31;
    const int kh    = lane >> 4;
    const int l16   = lane & 15;
    const int wv    = __builtin_amdgcn_readfirstlane((int)threadIdx.x) >> 5; // uniform wave id
    const int waveM = wv >> 2;                             // 0..1
    const int waveN = wv & 3;                              // 0..3

    const int nBlk = N >> 8;                               // 256-wide N tiles
    const int bm = blockIdx.x / nBlk;
    const int bn = blockIdx.x % nBlk;
    const long mBlk0 = (long)bm * A_ROWS;
    const long nBlk0 = (long)bn * B_ROWS;
    const long m0 = mBlk0 + waveM * 64;
    const long n0 = nBlk0 + waveN * 64;

    const unsigned ldsBase = (unsigned)(unsigned long long)(&smem[0]);
    const int KSTEPS = K >> 5;

    v8f acc[4][4] = {};

    // Prologue: stage 0 DMA (A tile + B tile) by wave 0.
    if (wv == 0) {
        tdm_load_tile(A + mBlk0 * K, ldsBase, A_ROWS, (unsigned)K);
        tdm_load_tile(W + nBlk0 * K, ldsBase + A_STAGE * 2, B_ROWS, (unsigned)K);
    }

    for (int it = 0; it < KSTEPS; ++it) {
        const int buf = it & 1;
        if (wv == 0) {
            if (it + 1 < KSTEPS) {
                const int nb = buf ^ 1;
                const unsigned base = ldsBase + (unsigned)nb * STAGE * 2;
                const long k1 = (long)(it + 1) * 32;
                tdm_load_tile(A + mBlk0 * K + k1, base, A_ROWS, (unsigned)K);
                tdm_load_tile(W + nBlk0 * K + k1, base + A_STAGE * 2, B_ROWS, (unsigned)K);
                __builtin_amdgcn_s_wait_tensorcnt(2);      // current stage done
            } else {
                __builtin_amdgcn_s_wait_tensorcnt(0);
            }
        }
        __syncthreads();                                   // stage visible to all waves

        const __bf16* lA = &smem[buf * STAGE];
        const __bf16* lB = &smem[buf * STAGE + A_STAGE];

        // B fragments: W[n, kh*16 .. +16) from padded LDS rows.
        v16bf bfr[4];
#pragma unroll
        for (int b = 0; b < 4; ++b) {
            const __bf16* pb = lB + (waveN * 64 + b * 16 + l16) * LDS_ROW + kh * 16;
            v8bf x0 = *(const v8bf*)pb;
            v8bf x1 = *(const v8bf*)(pb + 8);
#pragma unroll
            for (int i = 0; i < 8; ++i) { bfr[b][i] = x0[i]; bfr[b][i + 8] = x1[i]; }
        }
        // A fragments: A[m, kh*16 .. +16).
        v16bf afr[4];
#pragma unroll
        for (int a = 0; a < 4; ++a) {
            const __bf16* pa = lA + (waveM * 64 + a * 16 + l16) * LDS_ROW + kh * 16;
            v8bf lo = *(const v8bf*)pa;
            v8bf hi = *(const v8bf*)(pa + 8);
#pragma unroll
            for (int i = 0; i < 8; ++i) { afr[a][i] = lo[i]; afr[a][i + 8] = hi[i]; }
        }

#pragma unroll
        for (int a = 0; a < 4; ++a)
#pragma unroll
            for (int b = 0; b < 4; ++b)
                acc[a][b] = __builtin_amdgcn_wmma_f32_16x16x32_bf16(
                    false, afr[a], false, bfr[b], (short)0, acc[a][b], false, false);

        __syncthreads();                                   // protect buffer reuse
    }

    // Epilogue. C/D layout: lane l16 -> col n; VGPR r -> row r + 8*kh.
#pragma unroll
    for (int a = 0; a < 4; ++a) {
#pragma unroll
        for (int b = 0; b < 4; ++b) {
            const long n = n0 + b * 16 + l16;
#pragma unroll
            for (int r = 0; r < 8; ++r) {
                const long m = m0 + a * 16 + r + 8 * kh;
                float v = acc[a][b][r];
                if (MODE == 0) {
                    float t = v > 0.f ? v * v : 0.f;
                    outB[m * N + n] = (__bf16)t;
                } else if (MODE == 1) {
                    float s = 1.f / (1.f + __expf(-v));
                    outB[m * N + n] = (__bf16)s;
                } else {
                    outF[m * N + n] = v * (float)gate[m * N + n];
                }
            }
        }
    }
}

// ---------------------------------------------------------------------------
// Leaky integrate-and-fire scan (serial over S, one thread per (b,d) channel).
// ---------------------------------------------------------------------------
__global__ void lif_scan_kernel(const float* __restrict__ gated,
                                float* __restrict__ spikes,
                                int B, int S, int D) {
    int c = blockIdx.x * blockDim.x + threadIdx.x;
    if (c >= B * D) return;
    int b = c / D, d = c % D;
    const float* gp = gated + (long)b * S * D + d;
    float*       sp = spikes + (long)b * S * D + d;
    float mem = 0.f;
#pragma unroll 4
    for (int s = 0; s < S; ++s) {
        float inp   = gp[(long)s * D];
        float reset = (mem > THR_C) ? THR_C : 0.f;
        mem = BETA_C * mem + inp - reset;
        sp[(long)s * D] = (mem > THR_C) ? 1.f : 0.f;
    }
}

// ---------------------------------------------------------------------------
extern "C" void kernel_launch(void* const* d_in, const int* in_sizes, int n_in,
                              void* d_out, int out_size, void* d_ws, size_t ws_size,
                              hipStream_t stream) {
    const int B = 4, S = 4096, D = 1024;
    const int T = B * S;            // 16384 tokens
    const int N1 = 2 * D;           // 2048

    const float* x   = (const float*)d_in[0];
    const float* mix = (const float*)d_in[1];
    const float* WG  = (const float*)d_in[2];   // [2D, D]
    const float* WP  = (const float*)d_in[3];   // [D, D]
    const float* WS  = (const float*)d_in[4];   // [D, 2D]
    float* out = (float*)d_out;                 // [B,S,D] spikes f32

    char* ws = (char*)d_ws;
    size_t off = 0;
    auto carve = [&](size_t bytes) -> char* {
        char* p = ws + off;
        off = (off + bytes + 255) & ~(size_t)255;
        return p;
    };
    __bf16* xs_bf     = (__bf16*)carve((size_t)T * D * 2);     // 32 MB
    __bf16* wg_bf     = (__bf16*)carve((size_t)N1 * D * 2);    //  4 MB
    __bf16* wp_bf     = (__bf16*)carve((size_t)D * D * 2);     //  2 MB
    __bf16* ws_bf     = (__bf16*)carve((size_t)D * N1 * 2);    //  4 MB
    __bf16* growth_bf = (__bf16*)carve((size_t)T * N1 * 2);    // 64 MB
    __bf16* gate_bf   = (__bf16*)carve((size_t)T * D * 2);     // 32 MB
    float*  gated_f   = (float*)carve((size_t)T * D * 4);      // 64 MB
    (void)ws_size;

    // 1) token shift + bf16 convert
    {
        long total = (long)T * D;
        int blk = 256;
        prep_xs_kernel<<<(unsigned)((total + blk - 1) / blk), blk, 0, stream>>>(
            x, mix, xs_bf, S, D, total);
    }
    // 2) weight conversions
    {
        int blk = 256;
        long n;
        n = (long)N1 * D;
        f32_to_bf16_kernel<<<(unsigned)((n + blk - 1) / blk), blk, 0, stream>>>(WG, wg_bf, n);
        n = (long)D * D;
        f32_to_bf16_kernel<<<(unsigned)((n + blk - 1) / blk), blk, 0, stream>>>(WP, wp_bf, n);
        n = (long)D * N1;
        f32_to_bf16_kernel<<<(unsigned)((n + blk - 1) / blk), blk, 0, stream>>>(WS, ws_bf, n);
    }
    // 3) growth = relu(xs @ WG^T)^2            M=16384 N=2048 K=1024
    {
        int blocks = (T / 128) * (N1 / 256);
        wmma_gemm_kernel<0><<<blocks, 256, 0, stream>>>(
            xs_bf, wg_bf, T, N1, D, growth_bf, nullptr, nullptr);
    }
    // 4) gate = sigmoid(xs @ WP^T)             M=16384 N=1024 K=1024
    {
        int blocks = (T / 128) * (D / 256);
        wmma_gemm_kernel<1><<<blocks, 256, 0, stream>>>(
            xs_bf, wp_bf, T, D, D, gate_bf, nullptr, nullptr);
    }
    // 5) gated = (growth @ WS^T) * gate        M=16384 N=1024 K=2048
    {
        int blocks = (T / 128) * (D / 256);
        wmma_gemm_kernel<2><<<blocks, 256, 0, stream>>>(
            growth_bf, ws_bf, T, D, N1, nullptr, gated_f, gate_bf);
    }
    // 6) LIF scan -> spikes
    {
        int ch = B * D;
        lif_scan_kernel<<<(ch + 255) / 256, 256, 0, stream>>>(gated_f, out, B, S, D);
    }
}
